// ALIGNN_74861279969639
// MI455X (gfx1250) — matrix operand order repaired and checked by reading, not compiled, not run on hardware
//
#include <hip/hip_runtime.h>
#include <math.h>

// ---------------------------------------------------------------------------
// ALIGNN forward on MI455X (gfx1250, wave32).
// Dense layers -> v_wmma_f32_16x16x32_f16 tile GEMM:
//   - weights pre-packed to f16 in WMMA B-fragment order (4 node weights
//     fused into one N=1024 GEMM), fetched global->VGPR as b128 pairs
//   - activations register-prefetched + double-buffered into LDS in
//     A-fragment order (one barrier per K step; ds_load_b128 fragment reads)
//   - staging addressed as base+integer-offset so loads lower to
//     global_load_b128 (LOADcnt only), not flat_load (LOADcnt+DScnt)
// Message passing -> wave-per-edge kernels, contiguous 32B-per-lane rows
// (b128 merged loads), global fp32 atomics for segment sums, __shfl_xor
// wave reductions for LayerNorm.
// ---------------------------------------------------------------------------

typedef _Float16 vhalf16 __attribute__((ext_vector_type(16)));
typedef _Float16 vhalf4  __attribute__((ext_vector_type(4)));
typedef float    vfloat8 __attribute__((ext_vector_type(8)));
typedef float    vfloat4 __attribute__((ext_vector_type(4)));

#define HID 256
#define BM 64
#define BN 64
#define CHUNK 65536

__device__ __forceinline__ float wave_sum32(float v) {
  #pragma unroll
  for (int m = 16; m > 0; m >>= 1) v += __shfl_xor(v, m, 32);
  return v;
}
__device__ __forceinline__ float sigmoidf(float x) { return 1.0f / (1.0f + expf(-x)); }
__device__ __forceinline__ float siluf(float x)    { return x / (1.0f + expf(-x)); }

// -------------------- geometry / embedding helpers -------------------------

__global__ __launch_bounds__(256) void k_edge_geom(const float* __restrict__ r,
                                                   float* __restrict__ fcut,
                                                   float* __restrict__ feat80, int E) {
  int e = blockIdx.x * blockDim.x + threadIdx.x;
  if (e >= E) return;
  float x = r[3 * e], y = r[3 * e + 1], z = r[3 * e + 2];
  float d = sqrtf(x * x + y * y + z * z);
  const float ron2 = 7.5f * 7.5f, rc2 = 8.0f * 8.0f;
  float r2 = d * d;
  float s = ((rc2 - r2) * (rc2 - r2) * (rc2 + 2.0f * r2 - 3.0f * ron2)) /
            ((rc2 - ron2) * (rc2 - ron2) * (rc2 - ron2));
  fcut[e] = (d < 7.5f) ? 1.0f : ((d < 8.0f) ? s : 0.0f);
  const float gamma = 79.0f / 8.0f;
  #pragma unroll 4
  for (int i = 0; i < 80; i++) {
    float c = 8.0f * (float)i / 79.0f;
    float t = d - c;
    feat80[(size_t)e * 80 + i] = expf(-gamma * t * t);
  }
}

__global__ __launch_bounds__(256) void k_cos_rbf(const float* __restrict__ r,
                                                 const int* __restrict__ y_mask,
                                                 const int* __restrict__ lg_src,
                                                 const int* __restrict__ lg_dst,
                                                 float* __restrict__ feat40, int L) {
  int t = blockIdx.x * blockDim.x + threadIdx.x;
  if (t >= L) return;
  int e1 = y_mask[lg_src[t]], e2 = y_mask[lg_dst[t]];
  float ax = r[3 * e1], ay = r[3 * e1 + 1], az = r[3 * e1 + 2];
  float bx = r[3 * e2], by = r[3 * e2 + 1], bz = r[3 * e2 + 2];
  float dot = ax * bx + ay * by + az * bz;
  float na = sqrtf(ax * ax + ay * ay + az * az);
  float nb = sqrtf(bx * bx + by * by + bz * bz);
  float cosv = -dot / (na * nb);
  const float gamma = 39.0f / 2.0f;
  #pragma unroll 4
  for (int i = 0; i < 40; i++) {
    float c = -1.0f + 2.0f * (float)i / 39.0f;
    float d = cosv - c;
    feat40[(size_t)t * 40 + i] = expf(-gamma * d * d);
  }
}

__global__ __launch_bounds__(256) void k_gather_rows(const float* __restrict__ table,
                                                     const int* __restrict__ idx,
                                                     float* __restrict__ out,
                                                     int n, int w) {
  size_t total = (size_t)n * w;
  for (size_t i = (size_t)blockIdx.x * blockDim.x + threadIdx.x; i < total;
       i += (size_t)gridDim.x * blockDim.x) {
    size_t row = i / w, col = i - row * w;
    out[i] = table[(size_t)idx[row] * w + col];
  }
}

__global__ __launch_bounds__(256) void k_scatter_rows(const float* __restrict__ srcbuf,
                                                      const int* __restrict__ idx,
                                                      float* __restrict__ dst,
                                                      int n, int w) {
  size_t total = (size_t)n * w;
  for (size_t i = (size_t)blockIdx.x * blockDim.x + threadIdx.x; i < total;
       i += (size_t)gridDim.x * blockDim.x) {
    size_t row = i / w, col = i - row * w;
    dst[(size_t)idx[row] * w + col] = srcbuf[i];
  }
}

__global__ __launch_bounds__(256) void k_concat4(const float* __restrict__ a,
                                                 const float* __restrict__ b,
                                                 const float* __restrict__ c,
                                                 const float* __restrict__ d,
                                                 float* __restrict__ out) {
  int i = blockIdx.x * blockDim.x + threadIdx.x;
  if (i >= 1024) return;
  const float* p = (i < 256) ? a : (i < 512) ? b : (i < 768) ? c : d;
  out[i] = p[i & 255];
}

// -------------------- weight pre-pack (fp32 -> f16 B-fragment order) -------
// out[((ng*KT + kt)*32 + lane)*16 + e]; ng = n>>4, kt = k>>5, kl = k&31,
// hi = (kl>>4)&1, lane = (n&15)+16*hi, e = kl&15.  K zero-padded to KT*32.
__global__ __launch_bounds__(256) void k_pack_w(const float* __restrict__ W,
                                                _Float16* __restrict__ out,
                                                int K, int N, int KT) {
  size_t total = (size_t)KT * 32 * N;
  for (size_t i = (size_t)blockIdx.x * blockDim.x + threadIdx.x; i < total;
       i += (size_t)gridDim.x * blockDim.x) {
    int k = (int)(i / N);
    int n = (int)(i - (size_t)k * N);
    int kt = k >> 5, kl = k & 31;
    int hi = (kl >> 4) & 1;
    int lane = (n & 15) + (hi << 4);
    int e = kl & 15;
    int ng = n >> 4;
    float v = (k < K) ? W[(size_t)k * N + n] : 0.0f;
    out[(((size_t)ng * KT + kt) * 32 + lane) * 16 + e] = (_Float16)v;
  }
}

// -------------------- WMMA tile GEMM: C = A[M,K] * W[K,N] + bias -----------
// 256 threads = 8 wave32s; wave (wm,wn) owns a 16x32 slab of the 64x64 tile.
// Double-buffered LDS A staging: global loads for k-step kt+1 issue right
// after the barrier and drain while WMMAs for kt execute.
__global__ __launch_bounds__(256) void k_gemm_bias(const float* __restrict__ A,
                                                   const _Float16* __restrict__ Wp,
                                                   const float* __restrict__ bias,
                                                   float* __restrict__ C,
                                                   int M, int N, int K) {
  __shared__ _Float16 Ash[2][4][32][24] __attribute__((aligned(16)));
  const int tid  = threadIdx.x;
  const int lane = tid & 31;
  const int wave = tid >> 5;
  const int wm = (wave & 3) * 16;
  const int wn = (wave >> 2) * 32;
  const int tile_m = blockIdx.x * BM;
  const int tile_n = blockIdx.y * BN;
  const int KT = (K + 31) >> 5;
  const int hi  = lane >> 4;
  const int l15 = lane & 31 & 15;
  const size_t BUFS = 4u * 32u * 24u;  // halves per buffer

  // per-thread staging coordinates (2 units: 64 rows x 8 k-quads / 256 thr).
  // Integer offsets off the __restrict__ kernel arg keep these global_load.
  size_t aoff[2];
  _Float16* sp[2];
  #pragma unroll
  for (int j = 0; j < 2; j++) {
    int unit = tid + (j << 8);
    int row = unit >> 3;
    int k4  = (unit & 7) << 2;
    int gm = tile_m + row;
    gm = (gm < M) ? gm : (M - 1);                      // clamp (OOB masked at store)
    aoff[j] = (size_t)gm * K + k4;                     // K % 4 == 0 everywhere
    sp[j] = &Ash[0][row >> 4][(row & 15) + (((k4 >> 4) & 1) << 4)]
                [k4 & 15];
  }
  const size_t afrag = ((size_t)(wm >> 4) * 32 + lane) * 24;

  vfloat8 c0 = {}, c1 = {};
  vfloat4 f0 = *(const vfloat4*)(A + aoff[0]);          // prologue stage regs
  vfloat4 f1 = *(const vfloat4*)(A + aoff[1]);

  for (int kt = 0; kt < KT; kt++) {
    const size_t cb = (size_t)(kt & 1) * BUFS;
    vhalf4 h0, h1;
    h0[0] = (_Float16)f0[0]; h0[1] = (_Float16)f0[1];
    h0[2] = (_Float16)f0[2]; h0[3] = (_Float16)f0[3];
    h1[0] = (_Float16)f1[0]; h1[1] = (_Float16)f1[1];
    h1[2] = (_Float16)f1[2]; h1[3] = (_Float16)f1[3];
    *(vhalf4*)(sp[0] + cb) = h0;
    *(vhalf4*)(sp[1] + cb) = h1;
    __syncthreads();                                    // one barrier per K step
    if (kt + 1 < KT) {                                  // stage next slab into regs
      aoff[0] += 32; aoff[1] += 32;
      f0 = *(const vfloat4*)(A + aoff[0]);
      f1 = *(const vfloat4*)(A + aoff[1]);
      __builtin_prefetch(A + aoff[0] + 32, 0, 1);       // speculative, 2 slabs ahead
    }
    // A fragment: contiguous 32B per lane (ds_load_b128 x2)
    vhalf16 a = *(const vhalf16*)(&Ash[0][0][0][0] + cb + afrag);
    // B fragments: straight from packed weights (global_load_b128 x2 each)
    int ng0 = (tile_n + wn) >> 4;
    vhalf16 b0 = *(const vhalf16*)&Wp[(((size_t)ng0 * KT + kt) * 32 + lane) * 16];
    vhalf16 b1 = *(const vhalf16*)&Wp[((((size_t)ng0 + 1) * KT + kt) * 32 + lane) * 16];
    c0 = __builtin_amdgcn_wmma_f32_16x16x32_f16(false, a, false, b0, (short)0, c0, false, false);
    c1 = __builtin_amdgcn_wmma_f32_16x16x32_f16(false, a, false, b1, (short)0, c1, false, false);
  }

  int rbase = tile_m + wm + hi * 8;
  #pragma unroll
  for (int rr = 0; rr < 8; rr++) {
    int gr = rbase + rr;
    if (gr < M) {
      int gc0 = tile_n + wn + l15;
      int gc1 = gc0 + 16;
      C[(size_t)gr * N + gc0] = c0[rr] + bias[gc0];
      C[(size_t)gr * N + gc1] = c1[rr] + bias[gc1];
    }
  }
}

// -------------------- LayerNorm + SiLU (wave per row) ----------------------
template <int D>
__global__ __launch_bounds__(256) void k_ln_silu(float* __restrict__ X,
                                                 const float* __restrict__ g,
                                                 const float* __restrict__ b, int M) {
  const int lane = threadIdx.x & 31;
  int row = blockIdx.x * 8 + (threadIdx.x >> 5);
  if (row >= M) return;
  constexpr int D32 = D / 32;
  const int cb = lane * D32;               // contiguous per lane -> b128 merges
  float t[D32];
  float s = 0.0f;
  #pragma unroll
  for (int j = 0; j < D32; j++) { t[j] = X[(size_t)row * D + cb + j]; s += t[j]; }
  float mean = wave_sum32(s) / (float)D;
  float v = 0.0f;
  #pragma unroll
  for (int j = 0; j < D32; j++) { float d = t[j] - mean; v += d * d; }
  float rstd = rsqrtf(wave_sum32(v) / (float)D + 1e-5f);
  #pragma unroll
  for (int j = 0; j < D32; j++) {
    float u = (t[j] - mean) * rstd * g[cb + j] + b[cb + j];
    X[(size_t)row * D + cb + j] = siluf(u);
  }
}

// -------------------- edge-gated message kernel (wave per edge) ------------
// node4 fused columns: Ax = [0,256), Dx = [256,512), Bh = [512,768).
__global__ __launch_bounds__(256) void k_edge_msg(const float* __restrict__ node4,
                                                  const float* __restrict__ Ey,
                                                  const float* __restrict__ fcut,
                                                  const int* __restrict__ src,
                                                  const int* __restrict__ dst,
                                                  float* __restrict__ y,
                                                  float* __restrict__ ssh,
                                                  float* __restrict__ ss,
                                                  const float* __restrict__ ln_g,
                                                  const float* __restrict__ ln_b,
                                                  int skip_ln, int ne) {
  const int lane = threadIdx.x & 31;
  int e = blockIdx.x * 8 + (threadIdx.x >> 5);
  if (e >= ne) return;
  int s = src[e], d = dst[e];
  float fc = fcut ? fcut[e] : 1.0f;
  const int cb = lane * 8;                 // contiguous 32B per lane
  const float* axp = &node4[(size_t)s * 1024 + cb];
  const float* dxp = &node4[(size_t)d * 1024 + 256 + cb];
  const float* bhp = &node4[(size_t)s * 1024 + 512 + cb];
  const float* eyp = &Ey[(size_t)e * HID + cb];
  float m[8];
  float acc = 0.0f;
  #pragma unroll
  for (int j = 0; j < 8; j++) {
    float mm = axp[j] + dxp[j] + eyp[j];
    m[j] = mm;
    float sg = sigmoidf(mm) * fc;
    atomicAdd(&ssh[(size_t)d * HID + cb + j], sg * bhp[j]);
    atomicAdd(&ss[(size_t)d * HID + cb + j], sg);
    acc += mm;
  }
  float mean = wave_sum32(acc) / 256.0f;
  float v = 0.0f;
  #pragma unroll
  for (int j = 0; j < 8; j++) { float dd = m[j] - mean; v += dd * dd; }
  float rstd = rsqrtf(wave_sum32(v) / 256.0f + 1e-5f);
  #pragma unroll
  for (int j = 0; j < 8; j++) {
    float u = skip_ln ? m[j] : ((m[j] - mean) * rstd * ln_g[cb + j] + ln_b[cb + j]);
    y[(size_t)e * HID + cb + j] += siluf(u);
  }
}

// -------------------- node update (wave per node) --------------------------
// Du = node4 columns [768,1024).  x += silu(LN(Du + ssh/(ss+eps)))
__global__ __launch_bounds__(256) void k_node_update(float* __restrict__ x,
                                                     const float* __restrict__ node4,
                                                     const float* __restrict__ ssh,
                                                     const float* __restrict__ ss,
                                                     const float* __restrict__ g,
                                                     const float* __restrict__ b, int Nn) {
  const int lane = threadIdx.x & 31;
  int n = blockIdx.x * 8 + (threadIdx.x >> 5);
  if (n >= Nn) return;
  const int cb = lane * 8;
  float t[8];
  float acc = 0.0f;
  #pragma unroll
  for (int j = 0; j < 8; j++) {
    float h = ssh[(size_t)n * HID + cb + j] / (ss[(size_t)n * HID + cb + j] + 1e-6f);
    t[j] = node4[(size_t)n * 1024 + 768 + cb + j] + h;
    acc += t[j];
  }
  float mean = wave_sum32(acc) / 256.0f;
  float v = 0.0f;
  #pragma unroll
  for (int j = 0; j < 8; j++) { float dd = t[j] - mean; v += dd * dd; }
  float rstd = rsqrtf(wave_sum32(v) / 256.0f + 1e-5f);
  #pragma unroll
  for (int j = 0; j < 8; j++) {
    float u = (t[j] - mean) * rstd * g[cb + j] + b[cb + j];
    x[(size_t)n * HID + cb + j] += siluf(u);
  }
}

// -------------------- final FC + mean (wave per node) ----------------------
__global__ __launch_bounds__(256) void k_fc_mean(const float* __restrict__ x,
                                                 const float* __restrict__ w,
                                                 const float* __restrict__ b,
                                                 float* __restrict__ out, int Nn) {
  const int lane = threadIdx.x & 31;
  int n = blockIdx.x * 8 + (threadIdx.x >> 5);
  if (n >= Nn) return;
  const int cb = lane * 8;
  float acc = 0.0f;
  #pragma unroll
  for (int j = 0; j < 8; j++) acc += x[(size_t)n * HID + cb + j] * w[cb + j];
  float dot = wave_sum32(acc);
  if (lane == 0) atomicAdd(out, (dot + b[0]) / (float)Nn);
}

// ---------------------------------------------------------------------------
// Host orchestration. Param leaf order = jax tree_flatten (dicts key-sorted):
//   alignn[4]{edge,node} egg(14 leaves), angle_mlp1(4), angle_mlp2(4),
//   atom_mlp(4), atom_table(1), edge_mlp1(4), edge_mlp2(4), fc{b,w},
//   gcn[4] egg(14). egg: dst_gate{b,w}, dst_update{b,w}, edge_gate{b,w},
//   ln_edges{b,g}, ln_nodes{b,g}, src_gate{b,w}, src_update{b,w}.
//   mlp: lin{b,w}, ln{b,g}.
// Then: r=191, atomic_number=192, src=193, dst=194, y_mask=195,
//       lg_src=196, lg_dst=197.
// ---------------------------------------------------------------------------
extern "C" void kernel_launch(void* const* d_in, const int* in_sizes, int n_in,
                              void* d_out, int out_size, void* d_ws, size_t ws_size,
                              hipStream_t stream) {
  (void)n_in; (void)out_size; (void)ws_size;
  auto P = [&](int i) -> const float* { return (const float*)d_in[i]; };

  const int IDX_ANGLE1 = 112, IDX_ANGLE2 = 116, IDX_ATOM_MLP = 120, IDX_ATOM_TAB = 124;
  const int IDX_EDGE1 = 125, IDX_EDGE2 = 129, IDX_FC_B = 133, IDX_FC_W = 134;
  const int IDX_GCN = 135;
  const int IDX_R = 191;

  const float* r     = P(IDX_R);
  const int* atomno  = (const int*)d_in[IDX_R + 1];
  const int* src     = (const int*)d_in[IDX_R + 2];
  const int* dst     = (const int*)d_in[IDX_R + 3];
  const int* y_mask  = (const int*)d_in[IDX_R + 4];
  const int* lg_src  = (const int*)d_in[IDX_R + 5];
  const int* lg_dst  = (const int*)d_in[IDX_R + 6];

  const int N  = in_sizes[IDX_R + 1];
  const int E  = in_sizes[IDX_R + 2];
  const int ES = in_sizes[IDX_R + 4];
  const int LG = in_sizes[IDX_R + 5];

  // ---- workspace bump allocator ----
  size_t off = 0;
  auto alloc = [&](size_t nfloats) -> float* {
    float* p = (float*)((char*)d_ws + off);
    off += (nfloats * sizeof(float) + 255) & ~(size_t)255;
    return p;
  };
  size_t featN = (size_t)E * 80;
  if ((size_t)LG * 40 > featN) featN = (size_t)LG * 40;
  if ((size_t)N * 92  > featN) featN = (size_t)N * 92;
  size_t big = (E > LG) ? (size_t)E : (size_t)LG;
  int nodeMax = (N > ES) ? N : ES;

  float* fcut  = alloc(E);
  float* feat  = alloc(featN);
  float* tmp64 = alloc(big * 64);
  float* ybuf  = alloc((size_t)E * HID);
  float* zbuf  = alloc((size_t)LG * HID);
  float* xbuf  = alloc((size_t)N * HID);
  float* msub  = alloc((size_t)ES * HID);
  float* node4 = alloc((size_t)nodeMax * 1024);         // fused Ax|Dx|Bh|Du
  float* ssh   = alloc((size_t)nodeMax * HID);
  float* ss    = alloc((size_t)nodeMax * HID);
  float* Ey    = alloc((size_t)CHUNK * HID);
  float* bias4 = alloc(1024);
  _Float16* Wpack4 = (_Float16*)alloc(131072);          // 8*32*1024 halves
  _Float16* WpackE = (_Float16*)alloc(32768);           // 8*32*256 halves

  auto pack = [&](const float* W, _Float16* dstp, int K, int Ncols) {
    int KT = (K + 31) >> 5;
    size_t nelem = (size_t)KT * 32 * Ncols;
    int pblk = (int)((nelem + 255) / 256); if (pblk > 2048) pblk = 2048;
    k_pack_w<<<pblk, 256, 0, stream>>>(W, dstp, K, Ncols, KT);
  };
  auto gemm_run = [&](const float* A, const _Float16* Wp, const float* bias, float* C,
                      int M, int Ncols, int K) {
    dim3 grid((M + BM - 1) / BM, Ncols / BN);
    k_gemm_bias<<<grid, 256, 0, stream>>>(A, Wp, bias, C, M, Ncols, K);
  };
  auto gemm = [&](const float* A, const float* W, const float* bias, float* C,
                  int M, int Ncols, int K) {
    pack(W, Wpack4, K, Ncols);
    gemm_run(A, Wpack4, bias, C, M, Ncols, K);
  };
  auto ln64 = [&](float* X, const float* g, const float* b, int M) {
    k_ln_silu<64><<<(M + 7) / 8, 256, 0, stream>>>(X, g, b, M);
  };
  auto ln256 = [&](float* X, const float* g, const float* b, int M) {
    k_ln_silu<256><<<(M + 7) / 8, 256, 0, stream>>>(X, g, b, M);
  };

  auto egg = [&](int pb, const int* gsrc, const int* gdst, float* xf, float* yf,
                 const float* fc, int Nn, int Ne, int skip_ln) {
    // fused node GEMM: [src_gate | dst_gate | src_update | dst_update]
    const int KT = 8;                               // K = 256
    const size_t wstride = (size_t)16 * KT * 512;   // halves per 256-col weight
    pack(P(pb + 11), Wpack4 + 0 * wstride, HID, HID);   // src_gate.w
    pack(P(pb + 1),  Wpack4 + 1 * wstride, HID, HID);   // dst_gate.w
    pack(P(pb + 13), Wpack4 + 2 * wstride, HID, HID);   // src_update.w
    pack(P(pb + 3),  Wpack4 + 3 * wstride, HID, HID);   // dst_update.w
    k_concat4<<<4, 256, 0, stream>>>(P(pb + 10), P(pb + 0), P(pb + 12), P(pb + 2), bias4);
    gemm_run(xf, Wpack4, bias4, node4, Nn, 1024, HID);
    // edge gate weights packed once per egg (reused across chunks)
    pack(P(pb + 5), WpackE, HID, HID);
    hipMemsetAsync(ssh, 0, (size_t)Nn * HID * sizeof(float), stream);
    hipMemsetAsync(ss,  0, (size_t)Nn * HID * sizeof(float), stream);
    for (int eb = 0; eb < Ne; eb += CHUNK) {
      int ne = (Ne - eb < CHUNK) ? (Ne - eb) : CHUNK;
      gemm_run(yf + (size_t)eb * HID, WpackE, P(pb + 4), Ey, ne, HID, HID);
      k_edge_msg<<<(ne + 7) / 8, 256, 0, stream>>>(
          node4, Ey, fc ? fc + eb : nullptr, gsrc + eb, gdst + eb,
          yf + (size_t)eb * HID, ssh, ss, P(pb + 7), P(pb + 6), skip_ln, ne);
    }
    k_node_update<<<(Nn + 7) / 8, 256, 0, stream>>>(xf, node4, ssh, ss,
                                                    P(pb + 9), P(pb + 8), Nn);
  };

  // ---- embeddings ----
  k_edge_geom<<<(E + 255) / 256, 256, 0, stream>>>(r, fcut, feat, E);
  gemm(feat, P(IDX_EDGE1 + 1), P(IDX_EDGE1 + 0), tmp64, E, 64, 80);
  ln64(tmp64, P(IDX_EDGE1 + 3), P(IDX_EDGE1 + 2), E);
  gemm(tmp64, P(IDX_EDGE2 + 1), P(IDX_EDGE2 + 0), ybuf, E, HID, 64);
  ln256(ybuf, P(IDX_EDGE2 + 3), P(IDX_EDGE2 + 2), E);

  if (LG > 0) {
    k_cos_rbf<<<(LG + 255) / 256, 256, 0, stream>>>(r, y_mask, lg_src, lg_dst, feat, LG);
    gemm(feat, P(IDX_ANGLE1 + 1), P(IDX_ANGLE1 + 0), tmp64, LG, 64, 40);
    ln64(tmp64, P(IDX_ANGLE1 + 3), P(IDX_ANGLE1 + 2), LG);
    gemm(tmp64, P(IDX_ANGLE2 + 1), P(IDX_ANGLE2 + 0), zbuf, LG, HID, 64);
    ln256(zbuf, P(IDX_ANGLE2 + 3), P(IDX_ANGLE2 + 2), LG);
  }

  {
    int gblk = (int)(((size_t)N * 92 + 255) / 256); if (gblk > 4096) gblk = 4096;
    k_gather_rows<<<gblk, 256, 0, stream>>>(P(IDX_ATOM_TAB), atomno, feat, N, 92);
  }
  gemm(feat, P(IDX_ATOM_MLP + 1), P(IDX_ATOM_MLP + 0), xbuf, N, HID, 92);
  ln256(xbuf, P(IDX_ATOM_MLP + 3), P(IDX_ATOM_MLP + 2), N);

  // ---- 4 ALIGNN layers ----
  for (int l = 0; l < 4; l++) {
    int node_pb = l * 28 + 14;
    int edge_pb = l * 28;
    egg(node_pb, src, dst, xbuf, ybuf, fcut, N, E, 0);     // ybuf now holds m
    if (ES > 0 && LG > 0) {
      int gblk = (int)(((size_t)ES * HID + 255) / 256); if (gblk > 4096) gblk = 4096;
      k_gather_rows<<<gblk, 256, 0, stream>>>(ybuf, y_mask, msub, ES, HID);
      egg(edge_pb, lg_src, lg_dst, msub, zbuf, nullptr, ES, LG, (l == 3) ? 1 : 0);
      k_scatter_rows<<<gblk, 256, 0, stream>>>(msub, y_mask, ybuf, ES, HID);
    }
  }
  // ---- 4 GCN layers ----
  for (int g = 0; g < 4; g++) {
    egg(IDX_GCN + g * 14, src, dst, xbuf, ybuf, fcut, N, E, (g == 3) ? 1 : 0);
  }

  // ---- final FC + mean pooling ----
  hipMemsetAsync(d_out, 0, sizeof(float), stream);
  k_fc_mean<<<(N + 7) / 8, 256, 0, stream>>>(xbuf, P(IDX_FC_W), P(IDX_FC_B),
                                             (float*)d_out, N);
}
